// MemoryMHA_56229711839751
// MI455X (gfx1250) — matrix operand order repaired and checked by reading, not compiled, not run on hardware
//
#include <hip/hip_runtime.h>

// ---------------------------------------------------------------------------
// MemoryMHA on MI455X (gfx1250, wave32, WMMA + TDM).
// All GEMMs on v_wmma_f32_16x16x32_bf16 (bf16 in, f32 accumulate).
//  - weights pre-converted once to bf16 n-major (B-frag = two 16B loads)
//  - every GEMM processes TWO 16-wide output tiles per A-fragment (2x wmma
//    density per LDS read) and software-pipelines fragment loads so waits
//    land on older loads instead of serializing each WMMA.
//  - Q/head tile staging uses the Tensor Data Mover when available.
// ---------------------------------------------------------------------------

typedef __attribute__((ext_vector_type(16))) __bf16 v16bf;  // 8 VGPRs (A/B frag)
typedef __attribute__((ext_vector_type(8)))  __bf16 v8bf;   // 16 bytes
typedef __attribute__((ext_vector_type(8)))  float  v8f;    // 8 VGPRs (C/D frag)
typedef __attribute__((ext_vector_type(4)))  unsigned int ui4;
typedef __attribute__((ext_vector_type(8)))  int    i8v;
typedef __attribute__((ext_vector_type(4)))  int    i4v;

#define B_   128
#define N_   256
#define M_   8
#define D_   768
#define T_   264           // N + M
#define TP_  288           // T padded to a multiple of 32 (9 K-chunks)
#define KC_D (D_ / 32)     // 24 K-chunks over embed dim
#define KC_T (TP_ / 32)    // 9  K-chunks over key dim
#define WELEMS ((size_t)D_ * D_)

// ---- TDM availability -----------------------------------------------------
#define HAVE_TDM 0
#if defined(__has_builtin)
#  if __has_builtin(__builtin_amdgcn_tensor_load_to_lds)
#    undef HAVE_TDM
#    define HAVE_TDM 1
#  endif
#endif
#if defined(__has_include)
#  if __has_include(<hip/amd_detail/amd_gfx1250_TDM.h>)
#    define TDM_6ARG 1
#  else
#    define TDM_6ARG 0
#  endif
#else
#  define TDM_6ARG 0
#endif

#if HAVE_TDM
// TDM load of a 2D bf16 tile (rows x cols, row stride `stride` elems, dense
// in LDS). Issue from one lane (TDM ignores EXEC), then wait tensorcnt.
__device__ __forceinline__ void tdm_load_tile_bf16(unsigned int lds_addr,
                                                   const void* gptr,
                                                   int rows, int cols, int stride) {
  unsigned long long ga = (unsigned long long)(uintptr_t)gptr;
  ui4 g0;
  g0[0] = 1u;                                  // count=1 user descriptor
  g0[1] = lds_addr;                            // LDS byte address
  g0[2] = (unsigned int)ga;                    // global_addr[31:0]
  g0[3] = (unsigned int)((ga >> 32) & 0x1FFFFFFu) | (2u << 30);  // ga[56:32] | type=2
  i8v g1;
  g1[0] = (1 << 16);                           // data_size=1 (2B), no multicast/pad
  g1[1] = (cols & 0xFFFF) << 16;               // tensor_dim0[15:0]
  g1[2] = ((cols >> 16) & 0xFFFF) | ((rows & 0xFFFF) << 16);   // dim0 hi | dim1 lo
  g1[3] = ((rows >> 16) & 0xFFFF) | ((cols & 0xFFFF) << 16);   // dim1 hi | tile_dim0
  g1[4] = rows & 0xFFFF;                       // tile_dim1 (tile_dim2 = 0)
  g1[5] = stride;                              // tensor_dim0_stride[31:0]
  g1[6] = 0;
  g1[7] = 0;
  i4v gz = {0, 0, 0, 0};
#if TDM_6ARG
  i8v gz8 = {0, 0, 0, 0, 0, 0, 0, 0};
  __builtin_amdgcn_tensor_load_to_lds(g0, g1, gz, gz, gz8, 0);
#else
  __builtin_amdgcn_tensor_load_to_lds(g0, g1, gz, gz, 0);
#endif
}
#endif

// Stage a dense bf16 tile (contiguous in both global and LDS).
__device__ __forceinline__ void stage_tile_bf16(__bf16* lds, const __bf16* g,
                                                int nelems) {
#if HAVE_TDM
  if (threadIdx.x == 0)
    tdm_load_tile_bf16((unsigned int)(uintptr_t)lds, g, 16, nelems / 16, nelems / 16);
  __builtin_amdgcn_s_wait_tensorcnt(0);
#else
  for (int idx = threadIdx.x; idx < nelems / 8; idx += blockDim.x)
    ((v8bf*)lds)[idx] = ((const v8bf*)g)[idx];
#endif
  __syncthreads();
}

// ---- WMMA fragment loaders ------------------------------------------------
// A fragment (16x32 bf16) from a row-major [16 x stride] tile.
// lanes 0-15: row=lane, K {0..7,16..23}; lanes 16-31: K {8..15,24..31}.
__device__ __forceinline__ v16bf loadA(const __bf16* base, int stride) {
  int l   = threadIdx.x & 31;
  int row = l & 15;
  int kb  = (l < 16) ? 0 : 8;
  const __bf16* p = base + row * stride + kb;
  v8bf lo = *(const v8bf*)(p);
  v8bf hi = *(const v8bf*)(p + 16);
  return __builtin_shufflevector(lo, hi, 0,1,2,3,4,5,6,7,8,9,10,11,12,13,14,15);
}

// B fragment (32x16 bf16) from an "n-major" buffer: element(n,k) at base[n*stride+k].
// lanes 0-15: col=lane, K {0..7,16..23}; lanes 16-31: K {8..15,24..31}.
__device__ __forceinline__ v16bf loadB_nmajor(const __bf16* base, int stride) {
  int l  = threadIdx.x & 31;
  int n  = l & 15;
  int kb = (l < 16) ? 0 : 8;
  const __bf16* p = base + n * stride + kb;
  v8bf lo = *(const v8bf*)(p);
  v8bf hi = *(const v8bf*)(p + 16);
  return __builtin_shufflevector(lo, hi, 0,1,2,3,4,5,6,7,8,9,10,11,12,13,14,15);
}

#define WMMA_BF16(A, Bm, C) \
  __builtin_amdgcn_wmma_f32_16x16x32_bf16(false, (A), false, (Bm), (short)0, (C), false, false)

// Software-pipelined GEMM of one 16-row A panel against TWO 16-wide B panels.
// A row-major [16 x nchunks*32] (stride astride); B0/B1 n-major (stride bstride).
__device__ __forceinline__ void gemm_pair(const __bf16* Abase, int astride,
                                          const __bf16* B0, const __bf16* B1,
                                          int bstride, int nchunks,
                                          v8f& acc0, v8f& acc1) {
  v16bf a  = loadA(Abase, astride);
  v16bf b0 = loadB_nmajor(B0, bstride);
  v16bf b1 = loadB_nmajor(B1, bstride);
#pragma unroll 4
  for (int kc = 0; kc < nchunks - 1; ++kc) {
    v16bf an  = loadA(Abase + (kc + 1) * 32, astride);       // prefetch next chunk
    v16bf b0n = loadB_nmajor(B0 + (kc + 1) * 32, bstride);
    v16bf b1n = loadB_nmajor(B1 + (kc + 1) * 32, bstride);
    acc0 = WMMA_BF16(a, b0, acc0);
    acc1 = WMMA_BF16(a, b1, acc1);
    a = an; b0 = b0n; b1 = b1n;
  }
  acc0 = WMMA_BF16(a, b0, acc0);
  acc1 = WMMA_BF16(a, b1, acc1);
}

// ---------------------------------------------------------------------------
// Kernel 0: one-time weight prep.  Wt[w][n][k] (bf16) = W[w][k][n] (f32).
// ---------------------------------------------------------------------------
__global__ __launch_bounds__(256) void k_prep(
    const float* __restrict__ Wq, const float* __restrict__ Wk,
    const float* __restrict__ Wv, const float* __restrict__ Wo,
    __bf16* __restrict__ Wt) {
  const float* src[4] = {Wq, Wk, Wv, Wo};
  size_t total = 4 * WELEMS;
  for (size_t idx = (size_t)blockIdx.x * blockDim.x + threadIdx.x; idx < total;
       idx += (size_t)gridDim.x * blockDim.x) {
    size_t w = idx / WELEMS, r = idx % WELEMS;
    int n = (int)(r % D_);       // consecutive idx -> consecutive n: coalesced reads
    int k = (int)(r / D_);
    Wt[w * WELEMS + (size_t)n * D_ + k] = (__bf16)src[w][(size_t)k * D_ + n];
  }
}

// ---------------------------------------------------------------------------
// Kernel 1: q/k/v = cat @ W{q,k,v} + b.  cat = [x ; broadcast(memory) ; 0-pad].
// q -> [B,N,D] bf16, k -> [B,TP,D] bf16, v -> transposed vT [B,D,TP] bf16.
// 72 (weight, 32-wide d) pair-jobs over 8 waves.
// ---------------------------------------------------------------------------
__global__ __launch_bounds__(256) void k_qkv(
    const float* __restrict__ x, const float* __restrict__ mem,
    const __bf16* __restrict__ Wt,
    const float* __restrict__ bq, const float* __restrict__ bk,
    const float* __restrict__ bv,
    __bf16* __restrict__ qo, __bf16* __restrict__ ko, __bf16* __restrict__ vTo) {
  __shared__ __align__(16) __bf16 Acat[16 * D_];

  int b  = blockIdx.x / (TP_ / 16);
  int r0 = (blockIdx.x % (TP_ / 16)) * 16;

  // Stage cat tile (f32 -> bf16 conversion, so no TDM here).
  for (int idx = threadIdx.x; idx < 16 * D_; idx += blockDim.x) {
    int row = idx / D_, col = idx - row * D_;
    int t = r0 + row;
    float v = 0.f;
    if (t < N_)      v = x[((size_t)b * N_ + t) * D_ + col];
    else if (t < T_) v = mem[(size_t)(t - N_) * D_ + col];
    Acat[idx] = (__bf16)v;
  }
  __syncthreads();

  int wave = threadIdx.x >> 5;
  int l    = threadIdx.x & 31;
  int n    = l & 15;
  for (int pj = wave; pj < 3 * 24; pj += 8) {
    int which = pj / 24;                 // 0:q 1:k 2:v
    int d0    = (pj % 24) * 32;          // two 16-wide tiles: d0, d0+16
    const __bf16* W    = Wt + (size_t)which * WELEMS;
    const float*  bias = (which == 0) ? bq : (which == 1) ? bk : bv;

    v8f acc0 = {0.f, 0.f, 0.f, 0.f, 0.f, 0.f, 0.f, 0.f};
    v8f acc1 = acc0;
    gemm_pair(Acat, D_, W + (size_t)d0 * D_, W + (size_t)(d0 + 16) * D_, D_,
              KC_D, acc0, acc1);

    float bb0 = bias[d0 + n];
    float bb1 = bias[d0 + 16 + n];
#pragma unroll
    for (int i = 0; i < 8; ++i) {
      int   Mr = (l < 16) ? i : (8 + i);  // C/D layout: lanes 16-31 hold M=8..15
      int   t  = r0 + Mr;
      float v0 = acc0[i] + bb0;
      float v1 = acc1[i] + bb1;
      if (which == 0) {
        if (t < N_) {
          qo[((size_t)b * N_ + t) * D_ + d0 + n]      = (__bf16)v0;
          qo[((size_t)b * N_ + t) * D_ + d0 + 16 + n] = (__bf16)v1;
        }
      } else if (which == 1) {
        ko[((size_t)b * TP_ + t) * D_ + d0 + n]      = (__bf16)v0;
        ko[((size_t)b * TP_ + t) * D_ + d0 + 16 + n] = (__bf16)v1;
      } else {
        vTo[((size_t)b * D_ + d0 + n) * TP_ + t]      = (__bf16)v0;  // transposed
        vTo[((size_t)b * D_ + d0 + 16 + n) * TP_ + t] = (__bf16)v1;
      }
    }
  }
}

// ---------------------------------------------------------------------------
// Kernel 2: attention for one (batch, 16-query tile). Only query rows < N are
// launched, so the post-softmax memory-row mask is implicit. Padded key
// columns (>= 264) are excluded from softmax and forced to att = 0.
// ---------------------------------------------------------------------------
__global__ __launch_bounds__(256) void k_attn(
    const __bf16* __restrict__ q, const __bf16* __restrict__ k,
    const __bf16* __restrict__ vT, __bf16* __restrict__ head) {
  __shared__ __align__(16) __bf16 Qs[16 * D_];   // 24 KB
  __shared__ __align__(16) float  Ss[16 * TP_];  // 18 KB
  __shared__ __align__(16) __bf16 As[16 * TP_];  //  9 KB
  __shared__ float red[256];
  __shared__ float rowmax[16], rowsum[16];

  int b  = blockIdx.x / 16;
  int q0 = (blockIdx.x % 16) * 16;

  // Stage 16x768 Q tile via TDM (contiguous bf16 tile).
  stage_tile_bf16(Qs, q + ((size_t)b * N_ + q0) * D_, 16 * D_);

  int wave = threadIdx.x >> 5;
  int l    = threadIdx.x & 31;
  int n    = l & 15;
  const float scale = 0.03608439182435161f;  // 768^-0.5

  // --- scores = (Q K^T) * scale; 9 pairs of 16-key tiles over 8 waves ---
  for (int pr = wave; pr < TP_ / 32; pr += 8) {
    int n0 = pr * 32;
    v8f acc0 = {0.f, 0.f, 0.f, 0.f, 0.f, 0.f, 0.f, 0.f};
    v8f acc1 = acc0;
    const __bf16* kbase = k + ((size_t)b * TP_ + n0) * D_;  // n-major (key rows)
    gemm_pair(Qs, D_, kbase, kbase + (size_t)16 * D_, D_, KC_D, acc0, acc1);
#pragma unroll
    for (int i = 0; i < 8; ++i) {
      int Mr = (l < 16) ? i : (8 + i);
      Ss[Mr * TP_ + n0 + n]      = acc0[i] * scale;
      Ss[Mr * TP_ + n0 + 16 + n] = acc1[i] * scale;
    }
  }
  __syncthreads();

  // --- softmax over real keys (cols < 264); 16 threads per row ---
  int row = threadIdx.x >> 4;
  int sub = threadIdx.x & 15;
  float pm = -1e30f;
  for (int c = sub; c < T_; c += 16) pm = fmaxf(pm, Ss[row * TP_ + c]);
  red[threadIdx.x] = pm;
  __syncthreads();
  if (sub == 0) {
    float m = red[threadIdx.x];
    for (int i = 1; i < 16; ++i) m = fmaxf(m, red[threadIdx.x + i]);
    rowmax[row] = m;
  }
  __syncthreads();
  float m  = rowmax[row];
  float ps = 0.f;
  for (int c = sub; c < T_; c += 16) ps += __expf(Ss[row * TP_ + c] - m);
  red[threadIdx.x] = ps;
  __syncthreads();
  if (sub == 0) {
    float s = 0.f;
    for (int i = 0; i < 16; ++i) s += red[threadIdx.x + i];
    rowsum[row] = s;
  }
  __syncthreads();
  float inv = 1.f / rowsum[row];
  for (int c = sub; c < TP_; c += 16) {
    float a = (c < T_) ? __expf(Ss[row * TP_ + c] - m) * inv : 0.f;
    As[row * TP_ + c] = (__bf16)a;
  }
  __syncthreads();

  // --- head = att @ V; 24 pairs of 16-wide d tiles over 8 waves ---
  for (int pr = wave; pr < 24; pr += 8) {
    int d0 = pr * 32;
    v8f acc0 = {0.f, 0.f, 0.f, 0.f, 0.f, 0.f, 0.f, 0.f};
    v8f acc1 = acc0;
    const __bf16* vbase = vT + ((size_t)b * D_ + d0) * TP_;  // n-major in d
    gemm_pair(As, TP_, vbase, vbase + (size_t)16 * TP_, TP_, KC_T, acc0, acc1);
#pragma unroll
    for (int i = 0; i < 8; ++i) {
      int Mr = (l < 16) ? i : (8 + i);
      head[((size_t)b * N_ + q0 + Mr) * D_ + d0 + n]      = (__bf16)acc0[i];
      head[((size_t)b * N_ + q0 + Mr) * D_ + d0 + 16 + n] = (__bf16)acc1[i];
    }
  }
}

// ---------------------------------------------------------------------------
// Kernel 3: out = head @ Wo + bo  (f32 output).
// ---------------------------------------------------------------------------
__global__ __launch_bounds__(256) void k_oproj(
    const __bf16* __restrict__ head, const __bf16* __restrict__ WtO,
    const float* __restrict__ bo, float* __restrict__ out) {
  __shared__ __align__(16) __bf16 Ah[16 * D_];

  int b  = blockIdx.x / 16;
  int r0 = (blockIdx.x % 16) * 16;

  // Stage 16x768 head tile via TDM.
  stage_tile_bf16(Ah, head + ((size_t)b * N_ + r0) * D_, 16 * D_);

  int wave = threadIdx.x >> 5;
  int l    = threadIdx.x & 31;
  int n    = l & 15;
  for (int pr = wave; pr < 24; pr += 8) {
    int d0 = pr * 32;
    v8f acc0 = {0.f, 0.f, 0.f, 0.f, 0.f, 0.f, 0.f, 0.f};
    v8f acc1 = acc0;
    gemm_pair(Ah, D_, WtO + (size_t)d0 * D_, WtO + (size_t)(d0 + 16) * D_, D_,
              KC_D, acc0, acc1);
    float bb0 = bo[d0 + n];
    float bb1 = bo[d0 + 16 + n];
#pragma unroll
    for (int i = 0; i < 8; ++i) {
      int Mr = (l < 16) ? i : (8 + i);
      out[((size_t)b * N_ + r0 + Mr) * D_ + d0 + n]      = acc0[i] + bb0;
      out[((size_t)b * N_ + r0 + Mr) * D_ + d0 + 16 + n] = acc1[i] + bb1;
    }
  }
}

// ---------------------------------------------------------------------------
extern "C" void kernel_launch(void* const* d_in, const int* in_sizes, int n_in,
                              void* d_out, int out_size, void* d_ws, size_t ws_size,
                              hipStream_t stream) {
  const float* x   = (const float*)d_in[0];
  const float* mem = (const float*)d_in[1];
  const float* Wq  = (const float*)d_in[2];
  const float* bq  = (const float*)d_in[3];
  const float* Wk  = (const float*)d_in[4];
  const float* bk  = (const float*)d_in[5];
  const float* Wv  = (const float*)d_in[6];
  const float* bv  = (const float*)d_in[7];
  const float* Wo  = (const float*)d_in[8];
  const float* bo  = (const float*)d_in[9];
  float* out = (float*)d_out;

  // Workspace (bf16): Wt[4][D][D] | q [B,N,D] | k [B,TP,D] | vT [B,D,TP] | head [B,N,D]
  __bf16* Wt = (__bf16*)d_ws;
  __bf16* qo = Wt + 4 * WELEMS;
  __bf16* ko = qo + (size_t)B_ * N_ * D_;
  __bf16* vT = ko + (size_t)B_ * TP_ * D_;
  __bf16* hd = vT + (size_t)B_ * D_ * TP_;

  k_prep <<<512,              256, 0, stream>>>(Wq, Wk, Wv, Wo, Wt);
  k_qkv  <<<B_ * (TP_ / 16),  256, 0, stream>>>(x, mem, Wt, bq, bk, bv, qo, ko, vT);
  k_attn <<<B_ * 16,          256, 0, stream>>>(qo, ko, vT, hd);
  k_oproj<<<B_ * 16,          256, 0, stream>>>(hd, Wt + 3 * WELEMS, bo, out);
}
